// LSTMBlock_40707700031590
// MI455X (gfx1250) — matrix operand order repaired and checked by reading, not compiled, not run on hardware
//
#include <hip/hip_runtime.h>
#include <hip/hip_bf16.h>

// LSTM with per-timestep weights: B=64, S=64, IN=512, H=512 (4H=2048).
// Phase 0a: split x (fp32) -> bf16 hi/lo arrays (one-shot, parallel).
// Phase 0b: zero h0 hi/lo + grid-barrier counter.
// Phase 1 : Gx[t,b,:] = x[b,t,:] @ W_ih[t]^T for all t (parallel, WMMA).
// Phase 2 : ONE persistent kernel runs all 64 recurrent steps; 32 blocks sync
//           via a device-scope atomic grid barrier; c lives in LDS; h ping-pongs
//           through global memory as bf16 hi/lo; next-step weights are prefetched
//           (global_prefetch_b8) across the barrier to overlap sync with HBM fetch.
// fp32 accuracy on the bf16 WMMA pipe via exact Dekker hi/lo split (3 wmma/tile).

#define B_DIM 64
#define S_DIM 64
#define IN_DIM 512
#define H_DIM 512
#define G_DIM 2048  // 4*H
#define SEQ_BLOCKS (H_DIM / 16)  // 32 persistent blocks

typedef __attribute__((ext_vector_type(16))) __bf16 v16bf;
typedef __attribute__((ext_vector_type(8)))  unsigned int v8u;
typedef __attribute__((ext_vector_type(8)))  float v8f;

struct __align__(16) F4 { float x, y, z, w; };
struct __align__(16) U4 { unsigned int x, y, z, w; };

// Pack top-16-bits of two fp32 values into one dword (v_perm_b32).
__device__ __forceinline__ unsigned int pack_hi16(unsigned int u0, unsigned int u1) {
    return __builtin_amdgcn_perm(u1, u0, 0x07060302u);
}

// Exact Dekker split of an fp32 pair into packed bf16 hi dword and bf16 lo dword.
__device__ __forceinline__ void split_pair(float v0, float v1,
                                           unsigned int& hi01, unsigned int& lo01) {
    unsigned int u0 = __float_as_uint(v0);
    unsigned int u1 = __float_as_uint(v1);
    hi01 = pack_hi16(u0, u1);
    float f0 = __uint_as_float(u0 & 0xffff0000u);
    float f1 = __uint_as_float(u1 & 0xffff0000u);
    float l0 = v0 - f0;
    float l1 = v1 - f1;
    lo01 = pack_hi16(__float_as_uint(l0), __float_as_uint(l1));
}

// ---- A fragment (16x32 bf16) loaded directly from a pre-split bf16 array ----
__device__ __forceinline__ v16bf load_a_bf16(const unsigned short* a, int lda, int lane) {
    const int m  = lane & 15;
    const int kh = lane >> 4;
    const unsigned short* p = a + (size_t)m * lda + 8 * kh;
    U4 q0 = *(const U4*)(p);        // K offsets +0..+7
    U4 q1 = *(const U4*)(p + 16);   // K offsets +16..+23
    v8u r;
    r[0] = q0.x; r[1] = q0.y; r[2] = q0.z; r[3] = q0.w;
    r[4] = q1.x; r[5] = q1.y; r[6] = q1.z; r[7] = q1.w;
    return __builtin_bit_cast(v16bf, r);
}

// ---- B fragment (32x16 bf16) from fp32 weights with inline Dekker split ----
__device__ __forceinline__ void load_b_f32(const float* w, int ldw, int lane,
                                           v16bf& hi, v16bf& lo) {
    const int n  = lane & 15;
    const int kh = lane >> 4;
    const float* p = w + (size_t)n * ldw + 8 * kh;
    F4 q0 = *(const F4*)(p + 0);
    F4 q1 = *(const F4*)(p + 4);
    F4 q2 = *(const F4*)(p + 16);
    F4 q3 = *(const F4*)(p + 20);
    float v[16] = { q0.x,q0.y,q0.z,q0.w, q1.x,q1.y,q1.z,q1.w,
                    q2.x,q2.y,q2.z,q2.w, q3.x,q3.y,q3.z,q3.w };
    v8u uh, ul;
#pragma unroll
    for (int i = 0; i < 8; ++i) {
        unsigned int h01, l01;
        split_pair(v[2 * i], v[2 * i + 1], h01, l01);
        uh[i] = h01;
        ul[i] = l01;
    }
    hi = __builtin_bit_cast(v16bf, uh);
    lo = __builtin_bit_cast(v16bf, ul);
}

// One wave: 64(M) x 16(N) strip over K: acc += A*W^T (ahi*bhi + alo*bhi + ahi*blo).
__device__ __forceinline__ void gemm_strip(const unsigned short* Ahi,
                                           const unsigned short* Alo, int lda,
                                           const float* W, int ldw,
                                           int K, int lane, v8f acc[4]) {
    for (int k0 = 0; k0 < K; k0 += 32) {
        v16bf bhi, blo;
        load_b_f32(W + k0, ldw, lane, bhi, blo);
#pragma unroll
        for (int mt = 0; mt < 4; ++mt) {
            v16bf ahi = load_a_bf16(Ahi + (size_t)(16 * mt) * lda + k0, lda, lane);
            v16bf alo = load_a_bf16(Alo + (size_t)(16 * mt) * lda + k0, lda, lane);
            acc[mt] = __builtin_amdgcn_wmma_f32_16x16x32_bf16(
                false, ahi, false, bhi, (short)0, acc[mt], false, false);
            acc[mt] = __builtin_amdgcn_wmma_f32_16x16x32_bf16(
                false, alo, false, bhi, (short)0, acc[mt], false, false);
            acc[mt] = __builtin_amdgcn_wmma_f32_16x16x32_bf16(
                false, ahi, false, blo, (short)0, acc[mt], false, false);
        }
    }
}

// ---------------- Phase 0a: split x into bf16 hi/lo arrays ----------------
__global__ void split_x_kernel(const float* __restrict__ x,
                               unsigned short* __restrict__ xhi,
                               unsigned short* __restrict__ xlo) {
    int i = (blockIdx.x * blockDim.x + threadIdx.x) * 2;
    if (i < B_DIM * S_DIM * IN_DIM) {
        unsigned int h01, l01;
        split_pair(x[i], x[i + 1], h01, l01);
        *(unsigned int*)(xhi + i) = h01;
        *(unsigned int*)(xlo + i) = l01;
    }
}

// ---------------- Phase 0b: zero h0 (hi/lo) and the grid barrier counter ----------------
__global__ void lstm_init_kernel(unsigned short* __restrict__ hhi,
                                 unsigned short* __restrict__ hlo,
                                 unsigned int* __restrict__ bar) {
    int i = blockIdx.x * blockDim.x + threadIdx.x;
    if (i < B_DIM * H_DIM) { hhi[i] = 0; hlo[i] = 0; }
    if (i == 0) *bar = 0;
}

// ---------------- Phase 1: Gx[t,b,j] = sum_k x[b,t,k] * W_ih[t,j,k] ----------------
// grid = (G_DIM/128, S_DIM), block = 256 (8 waves); wave w covers 16 columns.
__global__ __launch_bounds__(256) void lstm_xw_kernel(const unsigned short* __restrict__ xhi,
                                                      const unsigned short* __restrict__ xlo,
                                                      const float* __restrict__ W_ih,
                                                      float* __restrict__ gx) {
    const int t    = blockIdx.y;
    const int lane = threadIdx.x & 31;
    const int j0   = blockIdx.x * 128 + (threadIdx.x >> 5) * 16;

    const unsigned short* Ahi = xhi + (size_t)t * IN_DIM;  // row stride S*IN
    const unsigned short* Alo = xlo + (size_t)t * IN_DIM;
    const float* W = W_ih + ((size_t)t * G_DIM + j0) * IN_DIM;

    v8f acc[4] = {};
    gemm_strip(Ahi, Alo, S_DIM * IN_DIM, W, IN_DIM, IN_DIM, lane, acc);

    const int n  = lane & 15;
    const int mo = (lane >> 4) * 8;
    float* dst = gx + (size_t)t * B_DIM * G_DIM + j0 + n;
#pragma unroll
    for (int mt = 0; mt < 4; ++mt)
#pragma unroll
        for (int r = 0; r < 8; ++r)
            dst[(size_t)(16 * mt + mo + r) * G_DIM] = acc[mt][r];
}

// ---------------- Phase 2: persistent kernel, all 64 steps ----------------
// grid = 32 blocks x 256 threads (8 waves). Block owns h-columns [jt, jt+16).
// Waves 0..3 = gates i,f,g,o over K [0,256); waves 4..7 = same gates, K [256,512).
// Gate halves reduce in LDS; c tile persists in LDS; grid barrier per step;
// next step's W_hh/Gx tiles are prefetched before the barrier.
__global__ __launch_bounds__(256) void lstm_seq_kernel(const float* __restrict__ gx,
                                                       const float* __restrict__ W_hh,
                                                       unsigned short* __restrict__ h_hi0,
                                                       unsigned short* __restrict__ h_lo0,
                                                       unsigned short* __restrict__ h_hi1,
                                                       unsigned short* __restrict__ h_lo1,
                                                       float* __restrict__ out,
                                                       unsigned int* __restrict__ bar) {
    __shared__ float gsm[8][B_DIM][16];
    __shared__ float csm[B_DIM][16];

    const int wave  = threadIdx.x >> 5;  // 0..7
    const int gate  = wave & 3;
    const int khalf = wave >> 2;         // 0/1 -> K split
    const int lane  = threadIdx.x & 31;
    const int jt    = blockIdx.x * 16;
    const int col   = gate * H_DIM + jt;
    const int kofs  = khalf * (H_DIM / 2);

    for (int i = threadIdx.x; i < B_DIM * 16; i += 256)
        csm[i >> 4][i & 15] = 0.0f;
    __syncthreads();

    for (int t = 0; t < S_DIM; ++t) {
        const unsigned short* Ahi = (t & 1) ? h_hi1 : h_hi0;
        const unsigned short* Alo = (t & 1) ? h_lo1 : h_lo0;
        unsigned short* Nhi = (t & 1) ? h_hi0 : h_hi1;
        unsigned short* Nlo = (t & 1) ? h_lo0 : h_lo1;

        const float* W = W_hh + ((size_t)t * G_DIM + col) * H_DIM + kofs;

        v8f acc[4] = {};
        gemm_strip(Ahi + kofs, Alo + kofs, H_DIM, W, H_DIM, H_DIM / 2, lane, acc);

        const int n  = lane & 15;
        const int mo = (lane >> 4) * 8;
#pragma unroll
        for (int mt = 0; mt < 4; ++mt)
#pragma unroll
            for (int r = 0; r < 8; ++r)
                gsm[wave][16 * mt + mo + r][n] = acc[mt][r];

        // ---- prefetch next step's weights (128KB/block) and Gx slice while we
        //      finish this step's pointwise work and sync (overlaps HBM latency) ----
        if (t + 1 < S_DIM) {
#pragma unroll
            for (int pi = 0; pi < 4; ++pi) {
                size_t off = ((size_t)pi * 256 + threadIdx.x) * 128;  // 4*256*128B = 128KB
                int gate_p = (int)(off >> 15);                        // 32KB per gate chunk
                size_t within = off & 0x7FFFu;
                const char* p = (const char*)(W_hh +
                    ((size_t)(t + 1) * G_DIM + (size_t)gate_p * H_DIM + jt) * H_DIM) + within;
                __builtin_prefetch(p, 0, 3);
            }
            {   // Gx[t+1] slice: 256 (b,gate) rows of 64B
                int b_p = threadIdx.x & 63;
                int g_p = threadIdx.x >> 6;
                const char* p = (const char*)(gx +
                    ((size_t)(t + 1) * B_DIM + b_p) * G_DIM + g_p * H_DIM + jt);
                __builtin_prefetch(p, 0, 3);
            }
        }
        __syncthreads();

        for (int i = threadIdx.x; i < B_DIM * 16; i += 256) {
            int b  = i >> 4;
            int nn = i & 15;
            const float* gsrc = gx + ((size_t)t * B_DIM + b) * G_DIM + jt + nn;
            float gi = gsm[0][b][nn] + gsm[4][b][nn] + gsrc[0 * H_DIM];
            float gf = gsm[1][b][nn] + gsm[5][b][nn] + gsrc[1 * H_DIM];
            float gc = gsm[2][b][nn] + gsm[6][b][nn] + gsrc[2 * H_DIM];
            float go = gsm[3][b][nn] + gsm[7][b][nn] + gsrc[3 * H_DIM];
            float ig = 1.0f / (1.0f + expf(-gi));
            float fg = 1.0f / (1.0f + expf(-gf));
            float gg = tanhf(gc);
            float og = 1.0f / (1.0f + expf(-go));
            float c = fg * csm[b][nn] + ig * gg;
            csm[b][nn] = c;
            float h = og * tanhf(c);
            out[((size_t)b * S_DIM + t) * H_DIM + jt + nn] = h;
            // next step's A operand in bf16 hi/lo
            unsigned int u = __float_as_uint(h);
            float hf = __uint_as_float(u & 0xffff0000u);
            float lr = h - hf;
            int cidx = b * H_DIM + jt + nn;
            Nhi[cidx] = (unsigned short)(u >> 16);
            Nlo[cidx] = (unsigned short)(__float_as_uint(lr) >> 16);
        }

        // ---- grid-wide barrier: every wave fences its own stores, then block signals ----
        __threadfence();      // wave-level release: h stores visible at device scope
        __syncthreads();
        if (threadIdx.x == 0) {
            __hip_atomic_fetch_add(bar, 1u, __ATOMIC_RELEASE, __HIP_MEMORY_SCOPE_AGENT);
            unsigned int target = (unsigned int)(t + 1) * SEQ_BLOCKS;
            while (__hip_atomic_load(bar, __ATOMIC_ACQUIRE, __HIP_MEMORY_SCOPE_AGENT) < target)
                __builtin_amdgcn_s_sleep(1);
        }
        __syncthreads();
    }
}

extern "C" void kernel_launch(void* const* d_in, const int* in_sizes, int n_in,
                              void* d_out, int out_size, void* d_ws, size_t ws_size,
                              hipStream_t stream) {
    const float* x    = (const float*)d_in[0];  // [B, S, IN]
    const float* W_ih = (const float*)d_in[1];  // [S, 4H, IN]
    const float* W_hh = (const float*)d_in[2];  // [S, 4H, H]
    float* out = (float*)d_out;                 // [B, S, H]

    // workspace layout (chunk sizes are multiples of 16 bytes):
    char* wsb = (char*)d_ws;
    float* gx = (float*)wsb;                                   // S*B*4H fp32 = 32 MB
    wsb += (size_t)S_DIM * B_DIM * G_DIM * sizeof(float);
    unsigned short* xhi = (unsigned short*)wsb;                // B*S*IN bf16
    wsb += (size_t)B_DIM * S_DIM * IN_DIM * sizeof(unsigned short);
    unsigned short* xlo = (unsigned short*)wsb;
    wsb += (size_t)B_DIM * S_DIM * IN_DIM * sizeof(unsigned short);
    unsigned short* h_hi[2];
    unsigned short* h_lo[2];
    for (int p = 0; p < 2; ++p) {
        h_hi[p] = (unsigned short*)wsb; wsb += (size_t)B_DIM * H_DIM * sizeof(unsigned short);
        h_lo[p] = (unsigned short*)wsb; wsb += (size_t)B_DIM * H_DIM * sizeof(unsigned short);
    }
    unsigned int* bar = (unsigned int*)wsb;

    split_x_kernel<<<(B_DIM * S_DIM * IN_DIM / 2 + 255) / 256, 256, 0, stream>>>(x, xhi, xlo);
    lstm_init_kernel<<<(B_DIM * H_DIM + 255) / 256, 256, 0, stream>>>(h_hi[0], h_lo[0], bar);

    lstm_xw_kernel<<<dim3(G_DIM / 128, S_DIM), 256, 0, stream>>>(xhi, xlo, W_ih, gx);

    lstm_seq_kernel<<<dim3(SEQ_BLOCKS), 256, 0, stream>>>(
        gx, W_hh, h_hi[0], h_lo[0], h_hi[1], h_lo[1], out, bar);
}
